// SimplifiedFraudDetectorGAT_87445534147095
// MI455X (gfx1250) — matrix operand order repaired and checked by reading, not compiled, not run on hardware
//
#include <hip/hip_runtime.h>
#include <hip/hip_bf16.h>

// ---------------------------------------------------------------------------
// SimplifiedFraudDetectorGAT on MI455X (gfx1250).
// All dense math on v_wmma_f32_16x16x32_bf16 (bf16 in, f32 accumulate).
// K is a compile-time template param (128 / 416) so the GEMM inner loop is
// branch-free: global_load_b128 -> v_cvt_pk_bf16_f32 -> v_wmma.
// combined@w1 never materialized: h1acc += per-block GEMMs (K=128 slices).
// Segmented softmax via ordered-uint atomicMax + f32 atomicAdd.
// ---------------------------------------------------------------------------

#define N_TYPES   11
#define N_ENT     50000
#define N_TX      131072
#define EDGES     131072
#define HID       128
#define TXF       394
#define PADK      416          // TXF padded up to a multiple of 32
#define NEG_SLOPE 0.2f

typedef __attribute__((ext_vector_type(16))) __bf16 v16bf;
typedef __attribute__((ext_vector_type(8)))  float  v8f;

// ----- order-preserving float<->uint encoding for atomicMax over floats -----
__device__ __forceinline__ unsigned f32_ord(float f) {
  unsigned u = __float_as_uint(f);
  return (u & 0x80000000u) ? ~u : (u | 0x80000000u);
}
__device__ __forceinline__ float ord_f32(unsigned o) {
  unsigned u = (o & 0x80000000u) ? (o & 0x7fffffffu) : ~o;
  return __uint_as_float(u);
}

// 16 consecutive f32 -> bf16 fragment (optional relu, row scale)
__device__ __forceinline__ v16bf frag_from_f32(float4 x0, float4 x1,
                                               float4 x2, float4 x3,
                                               float scale, int relu) {
  float f[16] = {x0.x, x0.y, x0.z, x0.w, x1.x, x1.y, x1.z, x1.w,
                 x2.x, x2.y, x2.z, x2.w, x3.x, x3.y, x3.z, x3.w};
  v16bf r;
#pragma unroll
  for (int e = 0; e < 16; ++e) {
    float v = f[e];
    if (relu) v = fmaxf(v, 0.0f);
    r[e] = (__bf16)(v * scale);
  }
  return r;
}

// ---------------------------------------------------------------------------
// bf16-WMMA GEMM: each wave computes a 16(M) x 64(N) strip (4 WMMA C tiles);
// 4 waves per block. K compile-time, multiple of 32 => branch-free hot loop.
//   Out[m,n] (op)= act( sum_k A'[m,k] * W[n, wcol0+k] + bias[n] )
//   A'[m,k] = scale_m * maybe_relu( A[rowmap(m)*lda + k] )
//   rowmap(m) = gather ? gather[m] : m      (edge src -> table row)
//   scale_m  = cnt ? 1/max(cnt[m],1) : 1    (scatter-mean divide)
// A fragment (16x32 bf16): lane m=lane&15; elems 0..7 -> k=k0+8*half+0..7,
// elems 8..15 -> k=k0+16+8*half+0..7.  B (32x16): lane n=lane&15; elem e ->
// k as for A, B[k][n]=W[n][k].  EXEC all-ones (full waves, no branches).
// ---------------------------------------------------------------------------
template <int K>
__global__ void wmma_gemm(const float* __restrict__ A, int lda,
                          const int* __restrict__ gather,
                          const float* __restrict__ cnt, int reluA,
                          const float* __restrict__ W, int ldw, int wcol0,
                          const float* __restrict__ bias,
                          float* __restrict__ Out, int ldo,
                          int accum, int reluOut) {
  constexpr int NT = 4;                       // N tiles per wave
  const int wv   = threadIdx.x >> 5;
  const int lane = threadIdx.x & 31;
  const int half = lane >> 4;
  const int lidx = lane & 15;
  const int m0 = (blockIdx.x * 4 + wv) << 4;
  const int n0 = blockIdx.y << 6;

  const int arow = m0 + lidx;
  const int amap = gather ? gather[arow] : arow;
  const float* ap = A + (long)amap * lda;
  const float ascale = cnt ? (1.0f / fmaxf(cnt[arow], 1.0f)) : 1.0f;

  const float* wp[NT];
#pragma unroll
  for (int t = 0; t < NT; ++t)
    wp[t] = W + (long)(n0 + 16 * t + lidx) * ldw + wcol0;

  v8f c[NT] = {};
#pragma unroll
  for (int k0 = 0; k0 < K; k0 += 32) {
    const float* a8 = ap + k0 + 8 * half;
    if (k0 + 32 < K)                          // compile-time per iteration
      __builtin_prefetch((const void*)(a8 + 32));
    v16bf afrag = frag_from_f32(*(const float4*)(a8 + 0),
                                *(const float4*)(a8 + 4),
                                *(const float4*)(a8 + 16),
                                *(const float4*)(a8 + 20),
                                ascale, reluA);
#pragma unroll
    for (int t = 0; t < NT; ++t) {
      const float* b16 = wp[t] + k0 + 8 * half;
      v16bf bfrag = frag_from_f32(*(const float4*)(b16 + 0),
                                  *(const float4*)(b16 + 4),
                                  *(const float4*)(b16 + 16),
                                  *(const float4*)(b16 + 20),
                                  1.0f, 0);
      c[t] = __builtin_amdgcn_wmma_f32_16x16x32_bf16(
          /*neg_a=*/false, afrag, /*neg_b=*/false, bfrag,
          /*c_mod=*/(short)0, c[t], /*reuse_a=*/false, /*reuse_b=*/false);
    }
  }

  // C layout: VGPR r -> M = r + 8*half; N = lane&15.
#pragma unroll
  for (int t = 0; t < NT; ++t) {
    const int col = n0 + 16 * t + lidx;
    const float bv = bias ? bias[col] : 0.0f;
#pragma unroll
    for (int r = 0; r < 8; ++r) {
      const int row = m0 + r + 8 * half;
      const long o = (long)row * ldo + col;
      float v = c[t][r] + bv;
      if (accum) v += Out[o];
      if (reluOut) v = fmaxf(v, 0.0f);
      Out[o] = v;
    }
  }
}

// ------------------------- pad / fill kernels ------------------------------
// dst[r*dcols + c] = (c < scols) ? src[r*scols + c] : 0
__global__ void pad_rows(const float* __restrict__ src, float* __restrict__ dst,
                         long rows, int scols, int dcols) {
  long i = (long)blockIdx.x * blockDim.x + threadIdx.x;
  if (i >= rows * (long)dcols) return;
  long r = i / dcols;
  int c = (int)(i - r * dcols);
  dst[i] = (c < scols) ? src[r * scols + c] : 0.0f;
}
__global__ void fill_f32(float* __restrict__ p, float v, long n) {
  long i = (long)blockIdx.x * blockDim.x + threadIdx.x;
  if (i < n) p[i] = v;
}
__global__ void fill_u32(unsigned* __restrict__ p, unsigned v, long n) {
  long i = (long)blockIdx.x * blockDim.x + threadIdx.x;
  if (i < n) p[i] = v;
}
// p[r*128 + c] = vec[c]
__global__ void fill_rows128(float* __restrict__ p,
                             const float* __restrict__ vec, long rows) {
  long i = (long)blockIdx.x * blockDim.x + threadIdx.x;
  if (i < rows * (long)HID) p[i] = vec[i & (HID - 1)];
}

// ----------------- attention dot products: wave per edge-row ---------------
__global__ void att_kernel(const float* __restrict__ xp,
                           const float* __restrict__ att_src,
                           const float* __restrict__ att_dst,
                           float* __restrict__ a_s, float* __restrict__ a_d) {
  long w = ((long)blockIdx.x * blockDim.x + threadIdx.x) >> 5;
  int lane = threadIdx.x & 31;
  if (w >= EDGES) return;
  const float4 x = *(const float4*)(xp + w * HID + lane * 4);
  const float4 s = *(const float4*)(att_src + lane * 4);
  const float4 d = *(const float4*)(att_dst + lane * 4);
  float ps = x.x * s.x + x.y * s.y + x.z * s.z + x.w * s.w;
  float pd = x.x * d.x + x.y * d.y + x.z * d.z + x.w * d.w;
#pragma unroll
  for (int off = 16; off > 0; off >>= 1) {
    ps += __shfl_xor(ps, off, 32);
    pd += __shfl_xor(pd, off, 32);
  }
  if (lane == 0) { a_s[w] = ps; a_d[w] = pd; }
}

__device__ __forceinline__ void msg_ids(long i, const int* src, const int* dst,
                                        int& sa, int& da) {
  if (i < EDGES) { sa = src[i]; da = dst[i]; }
  else           { sa = (int)(i - EDGES); da = sa; }   // add_self_loops part
}

// segment_max over dst_a via ordered-uint atomicMax
__global__ void segmax_kernel(const float* __restrict__ a_s,
                              const float* __restrict__ a_d,
                              const int* __restrict__ src,
                              const int* __restrict__ dst,
                              unsigned* __restrict__ mbits) {
  long i = (long)blockIdx.x * blockDim.x + threadIdx.x;
  if (i >= 2L * EDGES) return;
  int sa, da; msg_ids(i, src, dst, sa, da);
  float e = a_s[sa] + a_d[da];
  e = (e >= 0.0f) ? e : NEG_SLOPE * e;          // leaky_relu
  atomicMax(&mbits[da], f32_ord(e));
}

// ex = exp(e - m[seg]); denom[seg] += ex
__global__ void expdenom_kernel(const float* __restrict__ a_s,
                                const float* __restrict__ a_d,
                                const int* __restrict__ src,
                                const int* __restrict__ dst,
                                const unsigned* __restrict__ mbits,
                                float* __restrict__ ex,
                                float* __restrict__ denom) {
  long i = (long)blockIdx.x * blockDim.x + threadIdx.x;
  if (i >= 2L * EDGES) return;
  int sa, da; msg_ids(i, src, dst, sa, da);
  float e = a_s[sa] + a_d[da];
  e = (e >= 0.0f) ? e : NEG_SLOPE * e;
  float v = __expf(e - ord_f32(mbits[da]));
  ex[i] = v;
  atomicAdd(&denom[da], v);
}

// h[seg] += alpha * xp[srcrow]  — one wave per message, 4 columns per lane
__global__ void haccum_kernel(const float* __restrict__ ex,
                              const float* __restrict__ denom,
                              const float* __restrict__ xp,
                              const int* __restrict__ src,
                              const int* __restrict__ dst,
                              float* __restrict__ h) {
  long w = ((long)blockIdx.x * blockDim.x + threadIdx.x) >> 5;
  int lane = threadIdx.x & 31;
  if (w >= 2L * EDGES) return;
  int sa, da; msg_ids(w, src, dst, sa, da);
  float alpha = ex[w] / denom[da];
  const float4 x = *(const float4*)(xp + (long)sa * HID + lane * 4);
  float* hp = h + (long)da * HID + lane * 4;
  atomicAdd(hp + 0, alpha * x.x);
  atomicAdd(hp + 1, alpha * x.y);
  atomicAdd(hp + 2, alpha * x.z);
  atomicAdd(hp + 3, alpha * x.w);
}

// agg[dst[e]] += h[src[e]]; cnt[dst[e]] += 1  — one wave per edge
__global__ void agg_kernel(const float* __restrict__ h,
                           const int* __restrict__ src,
                           const int* __restrict__ dst,
                           float* __restrict__ agg,
                           float* __restrict__ cnt) {
  long w = ((long)blockIdx.x * blockDim.x + threadIdx.x) >> 5;
  int lane = threadIdx.x & 31;
  if (w >= EDGES) return;
  const int sa = src[w], da = dst[w];
  const float4 x = *(const float4*)(h + (long)sa * HID + lane * 4);
  float* apg = agg + (long)da * HID + lane * 4;
  atomicAdd(apg + 0, x.x);
  atomicAdd(apg + 1, x.y);
  atomicAdd(apg + 2, x.z);
  atomicAdd(apg + 3, x.w);
  if (lane == 0) atomicAdd(&cnt[da], 1.0f);
}

// logits = h2 @ w3.T + b3  — wave per row, 2 elems per lane over 64 cols
__global__ void logits_kernel(const float* __restrict__ h2,
                              const float* __restrict__ w3,
                              const float* __restrict__ b3,
                              float* __restrict__ out) {
  long w = ((long)blockIdx.x * blockDim.x + threadIdx.x) >> 5;
  int lane = threadIdx.x & 31;
  if (w >= N_TX) return;
  const float2 x  = *(const float2*)(h2 + w * 64 + lane * 2);
  const float2 wv = *(const float2*)(w3 + lane * 2);
  float p = x.x * wv.x + x.y * wv.y;
#pragma unroll
  for (int off = 16; off > 0; off >>= 1) p += __shfl_xor(p, off, 32);
  if (lane == 0) out[w] = p + b3[0];
}

// ---------------------------------------------------------------------------
extern "C" void kernel_launch(void* const* d_in, const int* in_sizes, int n_in,
                              void* d_out, int out_size, void* d_ws, size_t ws_size,
                              hipStream_t stream) {
  (void)in_sizes; (void)n_in; (void)out_size; (void)ws_size;
  const float* tx_x    = (const float*)d_in[0];   // [N_TX, TXF]
  const int*   eidx    = (const int*)  d_in[1];   // [N_TYPES, 2, EDGES]
  const float* ent     = (const float*)d_in[2];   // [N_TYPES, N_ENT, HID]
  const float* tx_w    = (const float*)d_in[3];   // [HID, TXF]
  const float* tx_b    = (const float*)d_in[4];   // [HID]
  const float* gat_w   = (const float*)d_in[5];   // [HID, HID]
  const float* att_src = (const float*)d_in[6];   // [HID]
  const float* att_dst = (const float*)d_in[7];   // [HID]
  const float* gat_b   = (const float*)d_in[8];   // [HID]
  const float* w1      = (const float*)d_in[9];   // [HID, 1536]
  const float* b1      = (const float*)d_in[10];  // [HID]
  const float* w2      = (const float*)d_in[11];  // [64, HID]
  const float* b2      = (const float*)d_in[12];  // [64]
  const float* w3      = (const float*)d_in[13];  // [1, 64]
  const float* b3      = (const float*)d_in[14];  // [1]
  float* out = (float*)d_out;

  // ---- workspace layout (regions time-shared across phases) --------------
  // [0, SZ_EH)            : tx_h (phase 1) -> xp (loop)
  // [SZ_EH, 2*SZ_EH)      : h1acc (live throughout)
  // [2*SZ_EH, +SZ_TPAD)   : tx_pad (phase 1)  ->  loop overlays:
  //                           hbuf (SZ_EH) | agg (SZ_EH) | small bufs | h2
  // [after big]           : txw_pad (tiny)
  const size_t SZ_EH   = (size_t)EDGES * HID * sizeof(float);   // 64 MiB
  const size_t SZ_TPAD = (size_t)N_TX * PADK * sizeof(float);   // 208 MiB
  char* base = (char*)d_ws;
  float* xp      = (float*)base;            // phase 1: tx_h
  float* tx_h    = xp;
  float* h1acc   = (float*)(base + SZ_EH);
  char*  big     = base + 2 * SZ_EH;
  float* tx_pad  = (float*)big;             // phase 1 only
  float* hbuf    = (float*)big;             // loop
  float* agg     = (float*)(big + SZ_EH);   // loop
  char*  sm      = big + 2 * SZ_EH;
  float*    cnt   = (float*)(sm + 0 * (size_t)N_TX * 4);
  float*    a_s   = (float*)(sm + 1 * (size_t)N_TX * 4);
  float*    a_d   = (float*)(sm + 2 * (size_t)N_TX * 4);
  unsigned* mbits = (unsigned*)(sm + 3 * (size_t)N_TX * 4);
  float*    denom = (float*)(sm + 4 * (size_t)N_TX * 4);
  float*    ex    = (float*)(sm + 5 * (size_t)N_TX * 4);        // 2E floats
  float*    h2    = (float*)(sm + 7 * (size_t)N_TX * 4);        // N_TX x 64
  float* txw_pad  = (float*)(big + SZ_TPAD);                    // [HID, PADK]

  const dim3 gblk(128, 1, 1);               // 4 waves per block
  auto blks = [](long n, int b) { return (unsigned)((n + b - 1) / b); };

  // 0) zero-pad tx_x and tx_w to K=416
  pad_rows<<<blks((long)N_TX * PADK, 256), 256, 0, stream>>>(
      tx_x, tx_pad, N_TX, TXF, PADK);
  pad_rows<<<blks((long)HID * PADK, 256), 256, 0, stream>>>(
      tx_w, txw_pad, HID, TXF, PADK);

  // 1) tx_h = relu(tx_pad @ txw_pad.T + tx_b)      [K=416]
  wmma_gemm<PADK><<<dim3(N_TX / 64, HID / 64), gblk, 0, stream>>>(
      tx_pad, PADK, nullptr, nullptr, 0, txw_pad, PADK, 0, tx_b,
      tx_h, HID, /*accum=*/0, /*reluOut=*/1);

  // 2) h1acc = b1 rows; h1acc += tx_h @ w1[:, 0:128].T
  fill_rows128<<<blks((long)N_TX * HID, 256), 256, 0, stream>>>(h1acc, b1, N_TX);
  wmma_gemm<HID><<<dim3(N_TX / 64, HID / 64), gblk, 0, stream>>>(
      tx_h, HID, nullptr, nullptr, 0, w1, HID * (1 + N_TYPES), 0, nullptr,
      h1acc, HID, /*accum=*/1, /*reluOut=*/0);

  // 3) per-type GAT message + fused h1acc accumulation
  for (int t = 0; t < N_TYPES; ++t) {
    const float* table = ent + (long)t * N_ENT * HID;
    const int* src = eidx + (long)(2 * t) * EDGES;
    const int* dst = src + EDGES;

    // xp = table[src] @ gat_w.T   (row-gather GEMM; tx_h dead now)
    wmma_gemm<HID><<<dim3(EDGES / 64, HID / 64), gblk, 0, stream>>>(
        table, HID, src, nullptr, 0, gat_w, HID, 0, nullptr,
        xp, HID, 0, 0);

    // a_s/a_d dot products
    att_kernel<<<blks((long)EDGES * 32, 256), 256, 0, stream>>>(
        xp, att_src, att_dst, a_s, a_d);

    // segmented softmax over dst_a (2E messages, E segments)
    fill_u32<<<blks(EDGES, 256), 256, 0, stream>>>(mbits, 0u, EDGES);
    segmax_kernel<<<blks(2L * EDGES, 256), 256, 0, stream>>>(
        a_s, a_d, src, dst, mbits);
    fill_f32<<<blks(EDGES, 256), 256, 0, stream>>>(denom, 0.0f, EDGES);
    expdenom_kernel<<<blks(2L * EDGES, 256), 256, 0, stream>>>(
        a_s, a_d, src, dst, mbits, ex, denom);

    // h = gat_b + segsum(alpha * xp[src_a])
    fill_rows128<<<blks((long)EDGES * HID, 256), 256, 0, stream>>>(
        hbuf, gat_b, EDGES);
    haccum_kernel<<<blks(2L * EDGES * 32, 256), 256, 0, stream>>>(
        ex, denom, xp, src, dst, hbuf);

    // scatter-mean numerator/denominator: agg[dst] += h[src]; cnt[dst] += 1
    fill_f32<<<blks((long)N_TX * HID, 256), 256, 0, stream>>>(
        agg, 0.0f, (long)N_TX * HID);
    fill_f32<<<blks(N_TX, 256), 256, 0, stream>>>(cnt, 0.0f, N_TX);
    agg_kernel<<<blks((long)EDGES * 32, 256), 256, 0, stream>>>(
        hbuf, src, dst, agg, cnt);

    // h1acc += (agg / clip(cnt,1)) @ w1[:, 128*(t+1):128*(t+2)].T
    wmma_gemm<HID><<<dim3(N_TX / 64, HID / 64), gblk, 0, stream>>>(
        agg, HID, nullptr, cnt, 0, w1, HID * (1 + N_TYPES), HID * (t + 1),
        nullptr, h1acc, HID, /*accum=*/1, /*reluOut=*/0);
  }

  // 4) h2 = relu( relu(h1acc) @ w2.T + b2 )   (h1 relu applied on A load)
  wmma_gemm<HID><<<dim3(N_TX / 64, 64 / 64), gblk, 0, stream>>>(
      h1acc, HID, nullptr, nullptr, /*reluA=*/1, w2, HID, 0, b2,
      h2, 64, 0, /*reluOut=*/1);

  // 5) logits = h2 @ w3.T + b3
  logits_kernel<<<blks((long)N_TX * 32, 256), 256, 0, stream>>>(h2, w3, b3, out);
}
